// MPNNblock_84894323573083
// MI455X (gfx1250) — matrix operand order repaired and checked by reading, not compiled, not run on hardware
//
#include <hip/hip_runtime.h>
#include <hip/hip_bf16.h>

typedef __attribute__((ext_vector_type(16))) _Float16 v16h;
typedef __attribute__((ext_vector_type(8)))  float    v8f;
typedef __attribute__((ext_vector_type(2)))  float    v2f;

#define N_NODES   50000
#define N_EDGES   800000
#define CH        128
#define EDGE_DIM  64

// ---- workspace layout (bytes) ----
#define H_BYTES    ((size_t)N_NODES * CH * 4)          // 25,600,000
#define OFF_H      ((size_t)0)
#define OFF_SUMS   (H_BYTES)
#define OFF_CNT    (2 * H_BYTES)                       // 51,200,000
#define CNT_BYTES  ((size_t)N_NODES * 4)               // 200,000
#define OFF_FRAGS  ((size_t)51400704)                  // 1KB aligned, past cnt
// fragment offsets in uints (each fragment = 32 lanes * 8 dwords = 256 uints)
#define FO_WE1 0
#define FO_WE2 (16 * 256)
#define FO_WN1 (48 * 256)
#define FO_WN2 (80 * 256)
#define FO_WM  (112 * 256)
#define FO_WU  (144 * 256)
#define N_FRAGS 176

// ---------------------------------------------------------------------------
// A-fragment gather: 16x32 f16 A-matrix layout (wave32):
//   lanes 0-15 : row m=lane,   VGPR j   packs K = kb+2j,   kb+2j+1   (j=0..3)
//                              VGPR 4+j packs K = kb+16+2j, +1       kb = kt*32
//   lanes 16-31: row m=lane-16, same but kb = kt*32 + 8
// rowp points at this lane's row (f32, contiguous K). NT variant streams
// (non-temporal) so read-once data does not churn L2.
// ---------------------------------------------------------------------------
template <bool NT>
__device__ __forceinline__ v16h load_afrag_f32(const float* rowp, int kt, int lane) {
  int kb = kt * 32 + ((lane & 16) ? 8 : 0);
  v16h a;
#pragma unroll
  for (int j = 0; j < 4; ++j) {
    const v2f* plo = (const v2f*)(rowp + kb + 2 * j);
    const v2f* phi = (const v2f*)(rowp + kb + 16 + 2 * j);
    v2f lo = NT ? __builtin_nontemporal_load(plo) : *plo;
    v2f hi = NT ? __builtin_nontemporal_load(phi) : *phi;
    a[2 * j]     = (_Float16)lo.x;
    a[2 * j + 1] = (_Float16)lo.y;
    a[8 + 2 * j]     = (_Float16)hi.x;
    a[8 + 2 * j + 1] = (_Float16)hi.y;
  }
  return a;
}

// B fragment: precomputed, per-lane contiguous 32 bytes (hot in WGP$/L2).
__device__ __forceinline__ v16h load_bfrag(const unsigned* frags, int kt, int nt, int lane) {
  return *(const v16h*)(frags + (((kt * 8 + nt) * 32 + lane) << 3));
}

__device__ __forceinline__ v8f wmma_f16(v16h a, v16h b, v8f c) {
  return __builtin_amdgcn_wmma_f32_16x16x32_f16(false, a, false, b, (short)0, c, false, false);
}

// One K-tile step of a 16x(16*8) GEMM: batch all 8 B-frag loads so they issue
// as one clause, then run 8 independent WMMAs back-to-back.
__device__ __forceinline__ void gemm_step(v8f acc[8], v16h a,
                                          const unsigned* fW, int kt, int lane) {
  v16h b[8];
#pragma unroll
  for (int nt = 0; nt < 8; ++nt) b[nt] = load_bfrag(fW, kt, nt, lane);
#pragma unroll
  for (int nt = 0; nt < 8; ++nt) acc[nt] = wmma_f16(a, b[nt], acc[nt]);
}

// ---------------------------------------------------------------------------
// Prep: convert weights (row-major [K,128] f32) into f16 B-fragment layout.
// B-frag layout: lane holds column n = nt*16 + lane%16,
// khalf = kt*32 + (lane<16 ? 0 : 8); dword j packs K = khalf+2j(+8 if j>=4), +1.
// ---------------------------------------------------------------------------
__global__ __launch_bounds__(128) void prep_frags_kernel(
    const float* __restrict__ We1, const float* __restrict__ We2,
    const float* __restrict__ Wn1, const float* __restrict__ Wn2,
    const float* __restrict__ Wm,  const float* __restrict__ Wu,
    unsigned* __restrict__ frags) {
  int gid = blockIdx.x * blockDim.x + threadIdx.x;
  if (gid >= N_FRAGS * 32) return;
  int f = gid >> 5, lane = gid & 31;
  const float* W; int lf; unsigned* dst;
  if (f < 16)       { W = We1; lf = f;       dst = frags + FO_WE1; }
  else if (f < 48)  { W = We2; lf = f - 16;  dst = frags + FO_WE2; }
  else if (f < 80)  { W = Wn1; lf = f - 48;  dst = frags + FO_WN1; }
  else if (f < 112) { W = Wn2; lf = f - 80;  dst = frags + FO_WN2; }
  else if (f < 144) { W = Wm;  lf = f - 112; dst = frags + FO_WM;  }
  else              { W = Wu;  lf = f - 144; dst = frags + FO_WU;  }
  int kt = lf >> 3, nt = lf & 7;
  int n = nt * 16 + (lane & 15);
  int khalf = kt * 32 + ((lane & 16) ? 8 : 0);
  unsigned* o = dst + ((lf * 32 + lane) << 3);
#pragma unroll
  for (int j = 0; j < 8; ++j) {
    int k = khalf + 2 * j + (j >= 4 ? 8 : 0);
    union { _Float16 h[2]; unsigned u; } pk;
    pk.h[0] = (_Float16)W[(size_t)k * CH + n];
    pk.h[1] = (_Float16)W[(size_t)(k + 1) * CH + n];
    o[j] = pk.u;
  }
}

// ---------------------------------------------------------------------------
// Node encoder: h = relu(x @ Wn1 + bn1) @ Wn2 + bn2     (50000 x 128)
// One wave = one 16-row tile, private 8KB LDS slice, no barriers needed.
// h stays regular-temporal: it fits in L2 (25.6MB) and is randomly gathered
// by the edge kernel.
// ---------------------------------------------------------------------------
__global__ __launch_bounds__(128) void node_encoder_kernel(
    const float* __restrict__ x, const unsigned* __restrict__ fWn1,
    const unsigned* __restrict__ fWn2, const float* __restrict__ bn1,
    const float* __restrict__ bn2, float* __restrict__ h) {
  __shared__ float lds[4][16 * CH];
  int lane = threadIdx.x & 31, wave = threadIdx.x >> 5;
  int tile = blockIdx.x * 4 + wave;
  if (tile >= N_NODES / 16) return;
  int rowbase = tile * 16;
  int m = lane & 15;
  int mhalf = (lane & 16) ? 8 : 0;
  float* tlds = lds[wave];

  // GEMM1: t = relu(x @ Wn1 + bn1)
  v8f acc[8] = {};
  const float* arow = x + (size_t)(rowbase + m) * CH;
#pragma unroll
  for (int kt = 0; kt < 4; ++kt)
    gemm_step(acc, load_afrag_f32<true>(arow, kt, lane), fWn1, kt, lane);
#pragma unroll
  for (int nt = 0; nt < 8; ++nt) {
    int n = nt * 16 + m;
    float b = bn1[n];
#pragma unroll
    for (int v = 0; v < 8; ++v)
      tlds[(mhalf + v) * CH + n] = fmaxf(acc[nt][v] + b, 0.0f);
  }

  // GEMM2: h = t @ Wn2 + bn2
#pragma unroll
  for (int nt = 0; nt < 8; ++nt) acc[nt] = (v8f){};
#pragma unroll
  for (int kt = 0; kt < 4; ++kt)
    gemm_step(acc, load_afrag_f32<false>(tlds + m * CH, kt, lane), fWn2, kt, lane);
#pragma unroll
  for (int nt = 0; nt < 8; ++nt) {
    int n = nt * 16 + m;
    float b = bn2[n];
#pragma unroll
    for (int v = 0; v < 8; ++v)
      h[(size_t)(rowbase + mhalf + v) * CH + n] = acc[nt][v] + b;
  }
}

// ---------------------------------------------------------------------------
// Edge pipeline (fused): e = relu(ea@We1+be1)@We2+be2 ; store e (streaming) ;
// m_in = e + h[col] ; msg = m_in@Wm+bm ; atomic scatter to sums[row], cnt[row].
// ---------------------------------------------------------------------------
__global__ __launch_bounds__(128) void edge_kernel(
    const float* __restrict__ edge_attr, const long long* __restrict__ edge_index,
    const float* __restrict__ h, const unsigned* __restrict__ fWe1,
    const unsigned* __restrict__ fWe2, const unsigned* __restrict__ fWm,
    const float* __restrict__ be1, const float* __restrict__ be2,
    const float* __restrict__ bm, float* __restrict__ e_out,
    float* __restrict__ sums, float* __restrict__ cnt) {
  __shared__ float lds[4][16 * CH];
  int lane = threadIdx.x & 31, wave = threadIdx.x >> 5;
  int tile = blockIdx.x * 4 + wave;
  if (tile >= N_EDGES / 16) return;
  int ebase = tile * 16;
  int m = lane & 15;
  int mhalf = (lane & 16) ? 8 : 0;
  float* tlds = lds[wave];
  const long long* rowIdx = edge_index;
  const long long* colIdx = edge_index + N_EDGES;

  // GEMM1: t1 = relu(edge_attr[16x64] @ We1 + be1), edge_attr streamed NT
  v8f acc[8] = {};
  const float* arow = edge_attr + (size_t)(ebase + m) * EDGE_DIM;
#pragma unroll
  for (int kt = 0; kt < 2; ++kt)
    gemm_step(acc, load_afrag_f32<true>(arow, kt, lane), fWe1, kt, lane);
#pragma unroll
  for (int nt = 0; nt < 8; ++nt) {
    int n = nt * 16 + m;
    float b = be1[n];
#pragma unroll
    for (int v = 0; v < 8; ++v)
      tlds[(mhalf + v) * CH + n] = fmaxf(acc[nt][v] + b, 0.0f);
  }

  // GEMM2: e = t1 @ We2 + be2  -> stream to global e output + keep in LDS
#pragma unroll
  for (int nt = 0; nt < 8; ++nt) acc[nt] = (v8f){};
#pragma unroll
  for (int kt = 0; kt < 4; ++kt)
    gemm_step(acc, load_afrag_f32<false>(tlds + m * CH, kt, lane), fWe2, kt, lane);
#pragma unroll
  for (int nt = 0; nt < 8; ++nt) {
    int n = nt * 16 + m;
    float b = be2[n];
#pragma unroll
    for (int v = 0; v < 8; ++v) {
      float val = acc[nt][v] + b;
      __builtin_nontemporal_store(val, e_out + (size_t)(ebase + mhalf + v) * CH + n);
      tlds[(mhalf + v) * CH + n] = val;
    }
  }

  // m_in = e + h[col] (cooperative gather within the wave; h is L2-resident)
  for (int r = 0; r < 16; ++r) {
    long long cix = colIdx[ebase + r];
    const float* hrow = h + (size_t)cix * CH;
#pragma unroll
    for (int j = 0; j < 4; ++j) {
      int n2 = j * 32 + lane;
      tlds[r * CH + n2] += hrow[n2];
    }
  }

  // GEMM3: msg = m_in @ Wm + bm
#pragma unroll
  for (int nt = 0; nt < 8; ++nt) acc[nt] = (v8f){};
#pragma unroll
  for (int kt = 0; kt < 4; ++kt)
    gemm_step(acc, load_afrag_f32<false>(tlds + m * CH, kt, lane), fWm, kt, lane);

  // Scatter-add msg to sums[row]; C-layout: lane holds col n=nt*16+m, rows mhalf+v.
  long long rid[8];
#pragma unroll
  for (int v = 0; v < 8; ++v) rid[v] = rowIdx[ebase + mhalf + v];
#pragma unroll
  for (int nt = 0; nt < 8; ++nt) {
    int n = nt * 16 + m;
    float b = bm[n];
#pragma unroll
    for (int v = 0; v < 8; ++v)
      atomicAdd(sums + (size_t)rid[v] * CH + n, acc[nt][v] + b);
  }
  if (lane < 16) atomicAdd(cnt + (int)rowIdx[ebase + lane], 1.0f);
}

// ---------------------------------------------------------------------------
// Update: out = (sums / max(cnt,1)) @ Wu + bu   (out streamed NT)
// ---------------------------------------------------------------------------
__global__ __launch_bounds__(128) void update_kernel(
    const float* __restrict__ sums, const float* __restrict__ cnt,
    const unsigned* __restrict__ fWu, const float* __restrict__ bu,
    float* __restrict__ out) {
  __shared__ float lds[4][16 * CH];
  int lane = threadIdx.x & 31, wave = threadIdx.x >> 5;
  int tile = blockIdx.x * 4 + wave;
  if (tile >= N_NODES / 16) return;
  int rowbase = tile * 16;
  int m = lane & 15;
  int mhalf = (lane & 16) ? 8 : 0;
  float* tlds = lds[wave];

  for (int r = 0; r < 16; ++r) {
    float inv = 1.0f / fmaxf(cnt[rowbase + r], 1.0f);
    const float* srow = sums + (size_t)(rowbase + r) * CH;
#pragma unroll
    for (int j = 0; j < 4; ++j) {
      int n2 = j * 32 + lane;
      tlds[r * CH + n2] = srow[n2] * inv;
    }
  }

  v8f acc[8] = {};
#pragma unroll
  for (int kt = 0; kt < 4; ++kt)
    gemm_step(acc, load_afrag_f32<false>(tlds + m * CH, kt, lane), fWu, kt, lane);
#pragma unroll
  for (int nt = 0; nt < 8; ++nt) {
    int n = nt * 16 + m;
    float b = bu[n];
#pragma unroll
    for (int v = 0; v < 8; ++v)
      __builtin_nontemporal_store(acc[nt][v] + b,
                                  out + (size_t)(rowbase + mhalf + v) * CH + n);
  }
}

// ---------------------------------------------------------------------------
extern "C" void kernel_launch(void* const* d_in, const int* in_sizes, int n_in,
                              void* d_out, int out_size, void* d_ws, size_t ws_size,
                              hipStream_t stream) {
  (void)in_sizes; (void)n_in; (void)out_size; (void)ws_size;
  const float*     x          = (const float*)d_in[0];
  const long long* edge_index = (const long long*)d_in[1];
  const float*     edge_attr  = (const float*)d_in[2];
  const float* We1 = (const float*)d_in[3];  const float* be1 = (const float*)d_in[4];
  const float* We2 = (const float*)d_in[5];  const float* be2 = (const float*)d_in[6];
  const float* Wn1 = (const float*)d_in[7];  const float* bn1 = (const float*)d_in[8];
  const float* Wn2 = (const float*)d_in[9];  const float* bn2 = (const float*)d_in[10];
  const float* Wm  = (const float*)d_in[11]; const float* bm  = (const float*)d_in[12];
  const float* Wu  = (const float*)d_in[13]; const float* bu  = (const float*)d_in[14];

  float* out   = (float*)d_out;                       // [50000*128]
  float* e_out = out + (size_t)N_NODES * CH;          // [800000*128]

  char* ws = (char*)d_ws;
  float*    h     = (float*)(ws + OFF_H);
  float*    sums  = (float*)(ws + OFF_SUMS);
  float*    cnt   = (float*)(ws + OFF_CNT);
  unsigned* frags = (unsigned*)(ws + OFF_FRAGS);

  // zero accumulators (sums + cnt are contiguous)
  (void)hipMemsetAsync(sums, 0, H_BYTES + CNT_BYTES, stream);

  prep_frags_kernel<<<(N_FRAGS * 32 + 127) / 128, 128, 0, stream>>>(
      We1, We2, Wn1, Wn2, Wm, Wu, frags);

  int node_tiles = N_NODES / 16;                      // 3125
  int edge_tiles = N_EDGES / 16;                      // 50000
  node_encoder_kernel<<<(node_tiles + 3) / 4, 128, 0, stream>>>(
      x, frags + FO_WN1, frags + FO_WN2, bn1, bn2, h);
  edge_kernel<<<(edge_tiles + 3) / 4, 128, 0, stream>>>(
      edge_attr, edge_index, h, frags + FO_WE1, frags + FO_WE2, frags + FO_WM,
      be1, be2, bm, e_out, sums, cnt);
  update_kernel<<<(node_tiles + 3) / 4, 128, 0, stream>>>(
      sums, cnt, frags + FO_WU, bu, out);
}
